// WaveNet_6502580486265
// MI455X (gfx1250) — matrix operand order repaired and checked by reading, not compiled, not run on hardware
//
#include <hip/hip_runtime.h>
#include <hip/hip_bf16.h>

// ---------------------------------------------------------------------------
// WaveNet on MI455X (gfx1250): all matmuls via V_WMMA_F32_16X16X32_BF16.
// Weights are pre-packed once per launch into fragment-ordered bf16 arrays in
// d_ws (lane's 16-element A-fragment = one contiguous 32B chunk), so the hot
// per-layer loop is: ds_load_b128 (B frags) + global_load_b128 (A frags) +
// v_wmma, with VALU only for the gated activation and activation bf16 cvt.
// Residual/skip intermediates live in d_ws in [B][T][32] layout; layers
// ping-pong residual buffers (safe for cross-tile dilated taps).
// ---------------------------------------------------------------------------

typedef __attribute__((ext_vector_type(16))) __bf16 v16bf;
typedef __attribute__((ext_vector_type(8)))  float  v8f;

#define T_LEN 16000
#define BATCH 8
#define NT    64             // tile width for start/head kernels
#define TILES (T_LEN / NT)   // 250
#define NTL   128            // tile width for layer kernels
#define TILESL (T_LEN / NTL) // 125

// Packed-fragment table (units of 512 bf16 = one 32-lane x 16-elem fragment):
//   Wstart: 16 frags   [mo*8 + ks]
//   Wdil  : 640 frags  [layer*16 + mtile*2 + kk]
//   Wres  : 320 frags  [layer*8 + mo*4 + ks]
//   We1   : 16 frags   [mtile]
//   We2   : 128 frags  [mtile*8 + ks]
#define FRAG_ELEMS 512
#define NFRAG_START 16
#define NFRAG_DIL   640
#define NFRAG_RES   320
#define NFRAG_E1    16
#define NFRAG_E2    128

// A-fragment K map (16-bit 16x32 A tile): lanes 0-15 carry
// K = 0..15, lanes 16-31 carry K = 16..31, M = lane&15.
__device__ __forceinline__ int akmap(int i, int lane) {
  return ((lane >> 4) << 4) + i;
}

// B-fragment loader: LDS holds the activation tile column-major [n][k]
// (k contiguous). Contiguous k order (same as akmap): lanes 0-15 hold K=0..15 of
// column N=lane, lanes 16-31 hold K=16..31 of column N=lane-16.
__device__ __forceinline__ v16bf ldsB(const __bf16* base, int kstride,
                                      int nbase, int k0, int lane) {
  const int col = nbase + (lane & 15);
  const int kb  = k0 + ((lane >> 4) << 4);
  return *(const v16bf*)(base + col * kstride + kb);
}

// Packed A-fragment loader: one aligned 32B chunk per lane.
__device__ __forceinline__ v16bf ldfragA(const __bf16* packed, int fragId,
                                         int lane) {
  return *(const v16bf*)(packed + ((size_t)fragId * 32 + lane) * 16);
}

__device__ __forceinline__ v8f wmma_bf16(v16bf a, v16bf b, v8f c) {
  return __builtin_amdgcn_wmma_f32_16x16x32_bf16(false, a, false, b,
                                                 (short)0, c, false, false);
}

// ---------------------------------------------------------------------------
// Weight packing: fp32 weights -> fragment-ordered bf16 in d_ws.
// dst[gid], gid = fragId*512 + lane*16 + i  (matches ldfragA).
// ---------------------------------------------------------------------------
__global__ __launch_bounds__(256) void wavenet_pack_kernel(
    const float* __restrict__ W, __bf16* __restrict__ dst, int mode,
    int total) {
  const int gid = blockIdx.x * 256 + threadIdx.x;
  if (gid >= total) return;
  const int i      = gid & 15;
  const int lane   = (gid >> 4) & 31;
  const int fragId = gid >> 9;
  const int mrow   = lane & 15;
  const int km     = akmap(i, lane);
  float v;
  switch (mode) {
    case 0: {  // Wstart [32][256]
      const int mo = fragId >> 3, ks = fragId & 7;
      v = W[(mo * 16 + mrow) * 256 + ks * 32 + km];
      break;
    }
    case 1: {  // Wdil [40][128][32][2]
      const int layer = fragId >> 4, mt = (fragId >> 1) & 7, kk = fragId & 1;
      v = W[(((size_t)layer * 128 + mt * 16 + mrow) * 32 + km) * 2 + kk];
      break;
    }
    case 2: {  // Wres [40][32][128]
      const int layer = fragId >> 3, mo = (fragId >> 2) & 1, ks = fragId & 3;
      v = W[((size_t)layer * 32 + mo * 16 + mrow) * 128 + ks * 32 + km];
      break;
    }
    case 3: {  // We1 [256][32]
      v = W[(fragId * 16 + mrow) * 32 + km];
      break;
    }
    default: {  // We2 [256][256]
      const int mt = fragId >> 3, ks = fragId & 7;
      v = W[(mt * 16 + mrow) * 256 + ks * 32 + km];
      break;
    }
  }
  dst[gid] = (__bf16)v;
}

// ---------------------------------------------------------------------------
// Start: residual0[b][t][o] = W_start[o][:] . x[b][:][t]; zero skip accum.
// ---------------------------------------------------------------------------
__global__ __launch_bounds__(256) void wavenet_start_kernel(
    const float* __restrict__ x,         // [B][256][T]
    const __bf16* __restrict__ fStart,   // packed fragments
    float* __restrict__ res_out,         // [B][T][32]
    float* __restrict__ skips) {         // [B][T][32] (zeroed here)
  __shared__ __attribute__((aligned(32))) __bf16 sX[NT * 256];  // [n][c] 32KB

  const int tid   = threadIdx.x;
  const int lane  = tid & 31;
  const int wave  = tid >> 5;
  const int b     = blockIdx.x / TILES;
  const int tbase = (blockIdx.x % TILES) * NT;

  {
    const float4* src =
        (const float4*)(x + ((size_t)b * 256 + tid) * T_LEN + tbase);
#pragma unroll
    for (int j = 0; j < NT / 4; ++j) {
      float4 v = src[j];
      sX[(j * 4 + 0) * 256 + tid] = (__bf16)v.x;
      sX[(j * 4 + 1) * 256 + tid] = (__bf16)v.y;
      sX[(j * 4 + 2) * 256 + tid] = (__bf16)v.z;
      sX[(j * 4 + 3) * 256 + tid] = (__bf16)v.w;
    }
  }
  __syncthreads();

  const int mo = wave >> 2;
  const int no = wave & 3;

  v8f c = {};
#pragma unroll
  for (int ks = 0; ks < 8; ++ks)
    c = wmma_bf16(ldfragA(fStart, mo * 8 + ks, lane),
                  ldsB(sX, 256, no * 16, ks * 32, lane), c);

#pragma unroll
  for (int r = 0; r < 8; ++r) {
    const int M = mo * 16 + ((lane >> 4) << 3) + r;
    const int N = no * 16 + (lane & 15);
    const size_t idx = ((size_t)b * T_LEN + tbase + N) * 32 + M;
    res_out[idx] = c[r];
    skips[idx]   = 0.0f;
  }
}

// ---------------------------------------------------------------------------
// One WaveNet layer (NTL=128 columns per workgroup):
//   h[128,NTL] = [Wd(:,:,0)|Wd(:,:,1)] (128x64) @ [R(t-d); R(t)] (64xNTL)
//   g = tanh(h)*sigmoid(h);  skip = Wres(32x128) @ g
//   res_out = res_in + skip; skips += skip
// ---------------------------------------------------------------------------
__global__ __launch_bounds__(256) void wavenet_layer_kernel(
    const float* __restrict__ res_in,   // [B][T][32]
    float* __restrict__ res_out,        // [B][T][32]
    float* __restrict__ skips,          // [B][T][32] accum
    const __bf16* __restrict__ fDil,    // this layer's conv fragments
    const __bf16* __restrict__ fRes,    // this layer's proj fragments
    int dilation) {
  __shared__ __attribute__((aligned(32))) __bf16 sB[NTL * 64];   // 16KB
  __shared__ __attribute__((aligned(32))) __bf16 sG[NTL * 128];  // 32KB

  const int tid   = threadIdx.x;
  const int lane  = tid & 31;
  const int wave  = tid >> 5;
  const int b     = blockIdx.x / TILESL;
  const int tbase = (blockIdx.x % TILESL) * NTL;

  // Stage: K rows 0..31 = channels at (t-d), 32..63 = channels at t.
  {
    const int col = tid >> 1;  // 0..127
    const int h   = tid & 1;
    const int t   = tbase + col;
    const int st  = h ? t : (t - dilation);
    __bf16* dst   = &sB[col * 64 + h * 32];
    if (st >= 0) {
      const float4* src =
          (const float4*)(res_in + ((size_t)b * T_LEN + st) * 32);
#pragma unroll
      for (int j = 0; j < 8; ++j) {
        float4 v = src[j];
        dst[j * 4 + 0] = (__bf16)v.x;
        dst[j * 4 + 1] = (__bf16)v.y;
        dst[j * 4 + 2] = (__bf16)v.z;
        dst[j * 4 + 3] = (__bf16)v.w;
      }
    } else {
#pragma unroll
      for (int j = 0; j < 32; ++j) dst[j] = (__bf16)0.0f;
    }
  }
  __syncthreads();

  // --- Dilated-conv matmul: wave owns M rows [wave*16, wave*16+16). ---
  const v16bf af0 = ldfragA(fDil, wave * 2 + 0, lane);  // taps at t-d
  const v16bf af1 = ldfragA(fDil, wave * 2 + 1, lane);  // taps at t

  v8f acc[8];
#pragma unroll
  for (int nt = 0; nt < 8; ++nt) {
    v8f c = {};
    c = wmma_bf16(af0, ldsB(sB, 64, nt * 16, 0, lane), c);
    c = wmma_bf16(af1, ldsB(sB, 64, nt * 16, 32, lane), c);
    acc[nt] = c;
  }

  // Gated activation -> bf16 -> LDS column-major [n][k=0..127].
#pragma unroll
  for (int nt = 0; nt < 8; ++nt) {
#pragma unroll
    for (int r = 0; r < 8; ++r) {
      const float h  = acc[nt][r];
      const float e  = __expf(-h);
      const float sg = 1.0f / (1.0f + e);
      const float e2 = e * e;
      const float th = (1.0f - e2) / (1.0f + e2);
      const int M = wave * 16 + ((lane >> 4) << 3) + r;
      const int N = nt * 16 + (lane & 15);
      sG[N * 128 + M] = (__bf16)(sg * th);
    }
  }
  __syncthreads();

  // --- Residual projection: skip = Wres(32x128) @ g. 2 tiles per wave. ---
  const int mo = wave >> 2;
  const int no = wave & 3;
  v16bf ar[4];
#pragma unroll
  for (int ks = 0; ks < 4; ++ks) ar[ks] = ldfragA(fRes, mo * 4 + ks, lane);

#pragma unroll
  for (int half = 0; half < 2; ++half) {
    const int nb = (no + half * 4) * 16;
    v8f c = {};
#pragma unroll
    for (int ks = 0; ks < 4; ++ks)
      c = wmma_bf16(ar[ks], ldsB(sG, 128, nb, ks * 32, lane), c);

#pragma unroll
    for (int r = 0; r < 8; ++r) {
      const int M = mo * 16 + ((lane >> 4) << 3) + r;  // channel
      const int N = nb + (lane & 15);                  // time
      const size_t idx = ((size_t)b * T_LEN + tbase + N) * 32 + M;
      const float s = c[r];
      res_out[idx] = res_in[idx] + s;
      skips[idx]   = skips[idx] + s;
    }
  }
}

// ---------------------------------------------------------------------------
// Head: h = relu(We1 @ relu(skips)); logits = We2 @ h  -> [B][256][T]
// ---------------------------------------------------------------------------
__global__ __launch_bounds__(256) void wavenet_head_kernel(
    const float* __restrict__ skips,   // [B][T][32]
    const __bf16* __restrict__ fE1,    // packed fragments
    const __bf16* __restrict__ fE2,    // packed fragments
    float* __restrict__ out) {         // [B][256][T]
  __shared__ __attribute__((aligned(32))) __bf16 sS[NT * 32];   //  4KB
  __shared__ __attribute__((aligned(32))) __bf16 sH[NT * 256];  // 32KB

  const int tid   = threadIdx.x;
  const int lane  = tid & 31;
  const int wave  = tid >> 5;
  const int b     = blockIdx.x / TILES;
  const int tbase = (blockIdx.x % TILES) * NT;

  {
    const int col = tid >> 2;
    const int q   = tid & 3;
    const float* src = skips + ((size_t)b * T_LEN + tbase + col) * 32 + q * 8;
    __bf16* dst = &sS[col * 32 + q * 8];
#pragma unroll
    for (int i = 0; i < 8; ++i) {
      float v = src[i];
      dst[i] = (__bf16)(v > 0.0f ? v : 0.0f);
    }
  }
  __syncthreads();

  // e1: wave handles m-tiles {wave, wave+8}; K=32 -> one WMMA per tile.
#pragma unroll
  for (int mt = 0; mt < 2; ++mt) {
    const int mtile = wave + mt * 8;
    const v16bf a = ldfragA(fE1, mtile, lane);
#pragma unroll
    for (int nt = 0; nt < 4; ++nt) {
      v8f c = {};
      c = wmma_bf16(a, ldsB(sS, 32, nt * 16, 0, lane), c);
#pragma unroll
      for (int r = 0; r < 8; ++r) {
        float v = c[r];
        v = v > 0.0f ? v : 0.0f;
        const int M = mtile * 16 + ((lane >> 4) << 3) + r;
        const int N = nt * 16 + (lane & 15);
        sH[N * 256 + M] = (__bf16)v;
      }
    }
  }
  __syncthreads();

  // e2: M=256, K=256 (8 k-steps).
#pragma unroll
  for (int mt = 0; mt < 2; ++mt) {
    const int mtile = wave + mt * 8;
    v16bf a2[8];
#pragma unroll
    for (int ks = 0; ks < 8; ++ks)
      a2[ks] = ldfragA(fE2, mtile * 8 + ks, lane);
#pragma unroll
    for (int nt = 0; nt < 4; ++nt) {
      v8f c = {};
#pragma unroll
      for (int ks = 0; ks < 8; ++ks)
        c = wmma_bf16(a2[ks], ldsB(sH, 256, nt * 16, ks * 32, lane), c);
#pragma unroll
      for (int r = 0; r < 8; ++r) {
        const int M = mtile * 16 + ((lane >> 4) << 3) + r;  // out channel
        const int N = tbase + nt * 16 + (lane & 15);        // time
        out[((size_t)b * 256 + M) * T_LEN + N] = c[r];
      }
    }
  }
}

// ---------------------------------------------------------------------------
extern "C" void kernel_launch(void* const* d_in, const int* in_sizes, int n_in,
                              void* d_out, int out_size, void* d_ws,
                              size_t ws_size, hipStream_t stream) {
  const float* x      = (const float*)d_in[0];  // [8,256,16000]
  const float* Wstart = (const float*)d_in[1];  // [32,256]
  const float* Wdil   = (const float*)d_in[2];  // [40,128,32,2]
  const float* Wres   = (const float*)d_in[3];  // [40,32,128]
  const float* We1    = (const float*)d_in[4];  // [256,32]
  const float* We2    = (const float*)d_in[5];  // [256,256]
  float* out = (float*)d_out;                   // [8,256,16000]

  const size_t resElems = (size_t)BATCH * T_LEN * 32;
  float* resA  = (float*)d_ws;
  float* resB  = resA + resElems;
  float* skips = resB + resElems;

  __bf16* fStart = (__bf16*)(skips + resElems);
  __bf16* fDil   = fStart + (size_t)NFRAG_START * FRAG_ELEMS;
  __bf16* fRes   = fDil + (size_t)NFRAG_DIL * FRAG_ELEMS;
  __bf16* fE1    = fRes + (size_t)NFRAG_RES * FRAG_ELEMS;
  __bf16* fE2    = fE1 + (size_t)NFRAG_E1 * FRAG_ELEMS;

  const dim3 block(256);

  // Pack all weights into fragment-ready bf16 (tiny; L2-resident afterwards).
  {
    const int tS = NFRAG_START * FRAG_ELEMS;
    const int tD = NFRAG_DIL * FRAG_ELEMS;
    const int tR = NFRAG_RES * FRAG_ELEMS;
    const int t1 = NFRAG_E1 * FRAG_ELEMS;
    const int t2 = NFRAG_E2 * FRAG_ELEMS;
    wavenet_pack_kernel<<<(tS + 255) / 256, block, 0, stream>>>(Wstart, fStart,
                                                                0, tS);
    wavenet_pack_kernel<<<(tD + 255) / 256, block, 0, stream>>>(Wdil, fDil, 1,
                                                                tD);
    wavenet_pack_kernel<<<(tR + 255) / 256, block, 0, stream>>>(Wres, fRes, 2,
                                                                tR);
    wavenet_pack_kernel<<<(t1 + 255) / 256, block, 0, stream>>>(We1, fE1, 3,
                                                                t1);
    wavenet_pack_kernel<<<(t2 + 255) / 256, block, 0, stream>>>(We2, fE2, 4,
                                                                t2);
  }

  wavenet_start_kernel<<<dim3(BATCH * TILES), block, 0, stream>>>(
      x, fStart, resA, skips);

  const float* rin = resA;
  float* rout = resB;
  for (int i = 0; i < 40; ++i) {
    const int d = 1 << (i % 10);
    wavenet_layer_kernel<<<dim3(BATCH * TILESL), block, 0, stream>>>(
        rin, rout, skips, fDil + (size_t)i * 16 * FRAG_ELEMS,
        fRes + (size_t)i * 8 * FRAG_ELEMS, d);
    const float* t = rout;
    rout = (float*)rin;
    rin = t;
  }

  wavenet_head_kernel<<<dim3(BATCH * TILES), block, 0, stream>>>(skips, fE1,
                                                                 fE2, out);
}
